// Qwen3MoEModel_87892210745579
// MI455X (gfx1250) — matrix operand order repaired and checked by reading, not compiled, not run on hardware
//
#include <hip/hip_runtime.h>
#include <hip/hip_bf16.h>
#include <math.h>

typedef __attribute__((ext_vector_type(16))) _Float16 v16h;
typedef __attribute__((ext_vector_type(8)))  _Float16 v8h;
typedef __attribute__((ext_vector_type(8)))  float    v8f;

constexpr int SEQ = 2048, DIM = 1024, NH = 16, HDIM = 64;
constexpr int NLAYER = 2, NEXP = 8, FFM = 768, VOCAB = 32000;
constexpr float EPS = 1e-6f;
constexpr int LDS_STRIDE = 40;   // halves per LDS row (80B rows -> 16B-aligned frags, bank-spread)

// ---------------------------------------------------------------- embed
__global__ void embed_kernel(const int* __restrict__ ids,
                             const float* __restrict__ emb,
                             float* __restrict__ h) {
    int t = blockIdx.x;
    long id = ids[t];
    for (int d = threadIdx.x; d < DIM; d += blockDim.x)
        h[(long)t * DIM + d] = emb[id * DIM + d];
}

// ---------------------------------------------------------------- rmsnorm (row-wise)
__global__ void rmsnorm_kernel(const float* __restrict__ in,
                               const float* __restrict__ w,
                               float* __restrict__ out, int dim) {
    __shared__ float red[256];
    int r = blockIdx.x;
    const float* x = in + (long)r * dim;
    float s = 0.f;
    for (int d = threadIdx.x; d < dim; d += 256) { float v = x[d]; s += v * v; }
    red[threadIdx.x] = s; __syncthreads();
    for (int st = 128; st > 0; st >>= 1) {
        if (threadIdx.x < st) red[threadIdx.x] += red[threadIdx.x + st];
        __syncthreads();
    }
    float rs = rsqrtf(red[0] / dim + EPS);
    for (int d = threadIdx.x; d < dim; d += 256)
        out[(long)r * dim + d] = w[d] * x[d] * rs;
}

// ---------------------------------------------------------------- weight prep: fp32 [K,N] -> f16 transposed [N,K]
__global__ void transpose_cvt_kernel(const float* __restrict__ src,
                                     _Float16* __restrict__ dst,
                                     int Kd, int Nc) {
    __shared__ _Float16 tile[32][33];
    int k0 = blockIdx.y * 32, n0 = blockIdx.x * 32;
    int tx = threadIdx.x, ty = threadIdx.y;          // 32 x 8
    for (int i = ty; i < 32; i += 8)
        tile[i][tx] = (_Float16)src[(long)(k0 + i) * Nc + n0 + tx];
    __syncthreads();
    for (int i = ty; i < 32; i += 8)
        dst[(long)(n0 + i) * Kd + k0 + tx] = tile[tx][i];
}

// ---------------------------------------------------------------- WMMA GEMM: C[M,N] = A[M,K] * Bt[N,K]^T
// A fp32 row-major; Bt f16, pre-transposed [N,K]. Block tile 256x128, 8 waves 4x2, wave 64x64.
__global__ void __launch_bounds__(256)
gemm_wmma(const float* __restrict__ A, const _Float16* __restrict__ Bt,
          float* __restrict__ C, int Nc, int Kd) {
    __shared__ __align__(16) _Float16 As[256 * LDS_STRIDE];  // 20 KB
    __shared__ __align__(16) _Float16 Bs[128 * LDS_STRIDE];  // 10 KB, Bs[n][k]

    int tid  = threadIdx.x;
    int lane = tid & 31, wave = tid >> 5;
    int wm = wave >> 1, wn = wave & 1;            // 4x2 wave grid
    int hi = lane >> 4, l15 = lane & 15;
    int rowBase = blockIdx.y * 256, colBase = blockIdx.x * 128;

    v8f acc[4][4] = {};

    for (int k0 = 0; k0 < Kd; k0 += 32) {
        // ---- stage B tile 128x32 (f16, already transposed): pure async copy to LDS
#pragma unroll
        for (int it = 0; it < 2; it++) {
            int c = it * 256 + tid;                  // 512 chunks of 16B
            int n = c >> 2, koff = (c & 3) * 8;      // 4 chunks per 32-half row
            const _Float16* gsrc = Bt + (long)(colBase + n) * Kd + k0 + koff;
            unsigned laddr = (unsigned)(size_t)(void*)(Bs + n * LDS_STRIDE + koff);
            asm volatile("global_load_async_to_lds_b128 %0, %1, off"
                         :: "v"(laddr), "v"(gsrc) : "memory");
        }
        // ---- stage A tile 256x32: fp32 load + packed cvt + LDS store
#pragma unroll
        for (int it = 0; it < 4; it++) {
            int gidx = it * 256 + tid;               // 1024 groups of 8 floats
            int row = gidx >> 2, kq = (gidx & 3) * 8;
            const float* src = A + (long)(rowBase + row) * Kd + k0 + kq;
            float4 f0 = *(const float4*)(src);
            float4 f1 = *(const float4*)(src + 4);
            v8h hp;
            hp[0] = (_Float16)f0.x; hp[1] = (_Float16)f0.y;
            hp[2] = (_Float16)f0.z; hp[3] = (_Float16)f0.w;
            hp[4] = (_Float16)f1.x; hp[5] = (_Float16)f1.y;
            hp[6] = (_Float16)f1.z; hp[7] = (_Float16)f1.w;
            *(v8h*)(As + row * LDS_STRIDE + kq) = hp;
        }
        // prefetch next K-tile while this one computes
        if (k0 + 32 < Kd) {
            __builtin_prefetch(A  + (long)(rowBase + (tid >> 2)) * Kd + k0 + 32, 0, 1);
            __builtin_prefetch(Bt + (long)(colBase + (tid >> 1)) * Kd + k0 + 32, 0, 1);
        }
        asm volatile("s_wait_asynccnt 0x0" ::: "memory");
        __syncthreads();

        // B fragments: lane col = l15, elem e -> K = hi*16 + e (contiguous 32B)
        v16h bfrag[4];
#pragma unroll
        for (int n = 0; n < 4; n++) {
            const _Float16* pb = Bs + (wn * 64 + n * 16 + l15) * LDS_STRIDE + hi * 16;
            ((v8h*)&bfrag[n])[0] = *(const v8h*)(pb);
            ((v8h*)&bfrag[n])[1] = *(const v8h*)(pb + 8);
        }
        // A fragments: lane row = l15; elem e -> K = hi*16 + e (contiguous 32B)
#pragma unroll
        for (int m = 0; m < 4; m++) {
            const _Float16* pa = As + (wm * 64 + m * 16 + l15) * LDS_STRIDE;
            v16h afrag;
            ((v8h*)&afrag)[0] = *(const v8h*)(pa + hi * 16);
            ((v8h*)&afrag)[1] = *(const v8h*)(pa + hi * 16 + 8);
#pragma unroll
            for (int n = 0; n < 4; n++)
                acc[m][n] = __builtin_amdgcn_wmma_f32_16x16x32_f16(
                    false, afrag, false, bfrag[n], (short)0, acc[m][n], false, false);
        }
        __syncthreads();
    }

    // ---- epilogue: 32-bit offsets (max index 2048*32000 < 2^31)
    unsigned uNc = (unsigned)Nc;
#pragma unroll
    for (int m = 0; m < 4; m++) {
        unsigned row0 = (unsigned)(rowBase + wm * 64 + m * 16 + 8 * hi);
#pragma unroll
        for (int n = 0; n < 4; n++) {
            unsigned base = row0 * uNc + (unsigned)(colBase + wn * 64 + n * 16 + l15);
#pragma unroll
            for (int r = 0; r < 8; r++)
                C[base + (unsigned)r * uNc] = acc[m][n][r];
        }
    }
}

// ---------------------------------------------------------------- per-head rmsnorm + rope (in-place)
__global__ void qknorm_rope_kernel(float* __restrict__ qk,
                                   const float* __restrict__ nw,
                                   const float* __restrict__ cosT,
                                   const float* __restrict__ sinT) {
    __shared__ float buf[64];
    __shared__ float xr[64];
    int t = blockIdx.x, hh = blockIdx.y, d = threadIdx.x;
    float* p = qk + ((long)t * NH + hh) * HDIM;
    float xv = p[d];
    buf[d] = xv * xv; __syncthreads();
    for (int s = 32; s > 0; s >>= 1) { if (d < s) buf[d] += buf[d + s]; __syncthreads(); }
    float rs = rsqrtf(buf[0] / HDIM + EPS);
    float xn = xv * rs * nw[d];
    xr[d] = xn; __syncthreads();
    float rot = (d < 32) ? -xr[d + 32] : xr[d - 32];
    p[d] = xn * cosT[(long)t * HDIM + d] + rot * sinT[(long)t * HDIM + d];
}

// ---------------------------------------------------------------- causal attention, online softmax (1 wave / (t,h))
__global__ void attention_kernel(const float* __restrict__ Q,
                                 const float* __restrict__ Kt,
                                 const float* __restrict__ Vt,
                                 float* __restrict__ O) {
    int t = blockIdx.x, hh = blockIdx.y, lane = threadIdx.x;
    const float scale = 0.125f;   // 1/sqrt(64)
    long base = (long)hh * HDIM;
    float q0 = Q[(long)t * DIM + base + lane];
    float q1 = Q[(long)t * DIM + base + lane + 32];
    float m = -1e30f, l = 0.f, a0 = 0.f, a1 = 0.f;
    for (int k = 0; k <= t; k++) {
        const float* kr = Kt + (long)k * DIM + base;
        float part = q0 * kr[lane] + q1 * kr[lane + 32];
        for (int off = 16; off > 0; off >>= 1) part += __shfl_xor(part, off, 32);
        float s = part * scale;
        float mn = fmaxf(m, s);
        float corr = __expf(m - mn);
        float p = __expf(s - mn);
        const float* vr = Vt + (long)k * DIM + base;
        l  = l * corr + p;
        a0 = a0 * corr + p * vr[lane];
        a1 = a1 * corr + p * vr[lane + 32];
        m = mn;
    }
    float inv = 1.f / l;
    O[(long)t * DIM + base + lane]      = a0 * inv;
    O[(long)t * DIM + base + lane + 32] = a1 * inv;
}

// ---------------------------------------------------------------- residual add
__global__ void add_kernel(float* __restrict__ a, const float* __restrict__ b, int n) {
    int i = blockIdx.x * 256 + threadIdx.x;
    if (i < n) a[i] += b[i];
}

// ---------------------------------------------------------------- router: x@gw -> softmax -> top2 -> renorm dense weights
__global__ void gate_topk_kernel(const float* __restrict__ x,
                                 const float* __restrict__ gw,
                                 float* __restrict__ dense) {
    __shared__ float red[128 * NEXP];
    int t = blockIdx.x, tid = threadIdx.x;
    float part[NEXP];
    for (int e = 0; e < NEXP; e++) part[e] = 0.f;
    for (int d = tid; d < DIM; d += 128) {
        float xv = x[(long)t * DIM + d];
        for (int e = 0; e < NEXP; e++) part[e] += xv * gw[(long)d * NEXP + e];
    }
    for (int e = 0; e < NEXP; e++) red[tid * NEXP + e] = part[e];
    __syncthreads();
    for (int s = 64; s > 0; s >>= 1) {
        if (tid < s)
            for (int e = 0; e < NEXP; e++) red[tid * NEXP + e] += red[(tid + s) * NEXP + e];
        __syncthreads();
    }
    if (tid == 0) {
        float mx = -1e30f;
        for (int e = 0; e < NEXP; e++) mx = fmaxf(mx, red[e]);
        float p[NEXP], sum = 0.f;
        for (int e = 0; e < NEXP; e++) { p[e] = __expf(red[e] - mx); sum += p[e]; }
        for (int e = 0; e < NEXP; e++) p[e] /= sum;
        int i1 = 0;
        for (int e = 1; e < NEXP; e++) if (p[e] > p[i1]) i1 = e;
        int i2 = (i1 == 0) ? 1 : 0;
        for (int e = 0; e < NEXP; e++) if (e != i1 && p[e] > p[i2]) i2 = e;
        float wsum = p[i1] + p[i2];
        for (int e = 0; e < NEXP; e++) dense[(long)t * NEXP + e] = 0.f;
        dense[(long)t * NEXP + i1] = p[i1] / wsum;
        dense[(long)t * NEXP + i2] = p[i2] / wsum;
    }
}

// ---------------------------------------------------------------- g = silu(g) * u
__global__ void silu_mul_kernel(float* __restrict__ g, const float* __restrict__ u, int n) {
    int i = blockIdx.x * 256 + threadIdx.x;
    if (i < n) { float x = g[i]; g[i] = (x / (1.f + __expf(-x))) * u[i]; }
}

// ---------------------------------------------------------------- h += dense[:,e] * y
__global__ void waccum_kernel(float* __restrict__ h, const float* __restrict__ y,
                              const float* __restrict__ dense, int e) {
    int i = blockIdx.x * 256 + threadIdx.x;
    if (i < SEQ * DIM) {
        int t = i / DIM;
        h[i] += dense[(long)t * NEXP + e] * y[i];
    }
}

// ================================================================ host orchestration
extern "C" void kernel_launch(void* const* d_in, const int* in_sizes, int n_in,
                              void* d_out, int out_size, void* d_ws, size_t ws_size,
                              hipStream_t stream) {
    (void)in_sizes; (void)n_in; (void)out_size; (void)ws_size;
    const int*   ids          = (const int*)  d_in[0];
    const float* emb          = (const float*)d_in[1];
    const float* wq           = (const float*)d_in[2];
    const float* wk           = (const float*)d_in[3];
    const float* wv           = (const float*)d_in[4];
    const float* wo           = (const float*)d_in[5];
    const float* q_norm_w     = (const float*)d_in[6];
    const float* k_norm_w     = (const float*)d_in[7];
    const float* attn_norm_w  = (const float*)d_in[8];
    const float* ffn_norm_w   = (const float*)d_in[9];
    const float* gate_w       = (const float*)d_in[10];
    const float* w_gate       = (const float*)d_in[11];
    const float* w_up         = (const float*)d_in[12];
    const float* w_down       = (const float*)d_in[13];
    const float* final_norm_w = (const float*)d_in[14];
    const float* out_w        = (const float*)d_in[15];
    const float* cosT         = (const float*)d_in[16];
    const float* sinT         = (const float*)d_in[17];
    float* out = (float*)d_out;
    float* ws  = (float*)d_ws;

    const long SD = (long)SEQ * DIM, SM = (long)SEQ * FFM;
    const long DD = (long)DIM * DIM, DM = (long)DIM * FFM;
    float* h     = ws;
    float* x     = ws + 1 * SD;
    float* q     = ws + 2 * SD;
    float* k     = ws + 3 * SD;
    float* v     = ws + 4 * SD;
    float* o     = ws + 5 * SD;
    float* y     = ws + 6 * SD;
    float* g     = ws + 7 * SD;
    float* u     = g + SM;
    float* dense = u + SM;
    // f16 transposed-weight region after fp32 scratch
    _Float16* hb   = (_Float16*)(dense + (long)SEQ * NEXP);
    _Float16* wqT  = hb;                      // [L][DIM][DIM]
    _Float16* wkT  = wqT  + NLAYER * DD;
    _Float16* wvT  = wkT  + NLAYER * DD;
    _Float16* woT  = wvT  + NLAYER * DD;
    _Float16* wgT  = woT  + NLAYER * DD;      // [L][E][FFM][DIM]
    _Float16* wuT  = wgT  + NLAYER * NEXP * DM;
    _Float16* wdT  = wuT  + NLAYER * NEXP * DM;  // [L][E][DIM][FFM]
    _Float16* outT = wdT  + NLAYER * NEXP * DM;  // [VOCAB][DIM]

    // ---- one-time weight convert+transpose (fp32 [K,N] -> f16 [N,K])
    dim3 tb(32, 8);
    for (int l = 0; l < NLAYER; l++) {
        transpose_cvt_kernel<<<dim3(DIM / 32, DIM / 32), tb, 0, stream>>>(wq + l * DD, wqT + l * DD, DIM, DIM);
        transpose_cvt_kernel<<<dim3(DIM / 32, DIM / 32), tb, 0, stream>>>(wk + l * DD, wkT + l * DD, DIM, DIM);
        transpose_cvt_kernel<<<dim3(DIM / 32, DIM / 32), tb, 0, stream>>>(wv + l * DD, wvT + l * DD, DIM, DIM);
        transpose_cvt_kernel<<<dim3(DIM / 32, DIM / 32), tb, 0, stream>>>(wo + l * DD, woT + l * DD, DIM, DIM);
        for (int e = 0; e < NEXP; e++) {
            long we = (long)(l * NEXP + e);
            transpose_cvt_kernel<<<dim3(FFM / 32, DIM / 32), tb, 0, stream>>>(w_gate + we * DM, wgT + we * DM, DIM, FFM);
            transpose_cvt_kernel<<<dim3(FFM / 32, DIM / 32), tb, 0, stream>>>(w_up   + we * DM, wuT + we * DM, DIM, FFM);
            transpose_cvt_kernel<<<dim3(DIM / 32, FFM / 32), tb, 0, stream>>>(w_down + we * DM, wdT + we * DM, FFM, DIM);
        }
    }
    transpose_cvt_kernel<<<dim3(VOCAB / 32, DIM / 32), tb, 0, stream>>>(out_w, outT, DIM, VOCAB);

    embed_kernel<<<SEQ, 256, 0, stream>>>(ids, emb, h);

    dim3 gD(DIM / 128, SEQ / 256);     // 2048x1024 output
    dim3 gM(FFM / 128, SEQ / 256);     // 2048x768 output
    dim3 gV(VOCAB / 128, SEQ / 256);   // 2048x32000 output
    dim3 gTH(SEQ, NH);
    int nSD = SEQ * DIM, nSM = SEQ * FFM;

    for (int l = 0; l < NLAYER; l++) {
        rmsnorm_kernel<<<SEQ, 256, 0, stream>>>(h, attn_norm_w + (long)l * DIM, x, DIM);

        gemm_wmma<<<gD, 256, 0, stream>>>(x, wqT + l * DD, q, DIM, DIM);
        gemm_wmma<<<gD, 256, 0, stream>>>(x, wkT + l * DD, k, DIM, DIM);
        gemm_wmma<<<gD, 256, 0, stream>>>(x, wvT + l * DD, v, DIM, DIM);

        qknorm_rope_kernel<<<gTH, 64, 0, stream>>>(q, q_norm_w + (long)l * HDIM, cosT, sinT);
        qknorm_rope_kernel<<<gTH, 64, 0, stream>>>(k, k_norm_w + (long)l * HDIM, cosT, sinT);

        attention_kernel<<<gTH, 32, 0, stream>>>(q, k, v, o);

        gemm_wmma<<<gD, 256, 0, stream>>>(o, woT + l * DD, y, DIM, DIM);
        add_kernel<<<nSD / 256, 256, 0, stream>>>(h, y, nSD);

        rmsnorm_kernel<<<SEQ, 256, 0, stream>>>(h, ffn_norm_w + (long)l * DIM, x, DIM);
        gate_topk_kernel<<<SEQ, 128, 0, stream>>>(x, gate_w + (long)l * DIM * NEXP, dense);

        for (int e = 0; e < NEXP; e++) {
            long we = (long)(l * NEXP + e);
            gemm_wmma<<<gM, 256, 0, stream>>>(x, wgT + we * DM, g, FFM, DIM);
            gemm_wmma<<<gM, 256, 0, stream>>>(x, wuT + we * DM, u, FFM, DIM);
            silu_mul_kernel<<<nSM / 256, 256, 0, stream>>>(g, u, nSM);
            gemm_wmma<<<gD, 256, 0, stream>>>(g, wdT + we * DM, y, DIM, FFM);
            waccum_kernel<<<nSD / 256, 256, 0, stream>>>(h, y, dense, e);
        }
    }

    rmsnorm_kernel<<<SEQ, 256, 0, stream>>>(h, final_norm_w, x, DIM);
    gemm_wmma<<<gV, 256, 0, stream>>>(x, outT, out, VOCAB, DIM);
}
